// Multi_Head_Attention_14894946583254
// MI455X (gfx1250) — matrix operand order repaired and checked by reading, not compiled, not run on hardware
//
#include <hip/hip_runtime.h>

typedef unsigned short u16;
typedef __attribute__((ext_vector_type(16))) __bf16          v16bf;
typedef __attribute__((ext_vector_type(8)))  float           v8f;
typedef __attribute__((ext_vector_type(8)))  unsigned short  v8u16;

union BF16x16 {
  v16bf v;
  v8u16 h[2];
};

__device__ __forceinline__ u16 f2bf(float f) {
  unsigned u = __builtin_bit_cast(unsigned, f);
  u += 0x7FFFu + ((u >> 16) & 1u);           // round-to-nearest-even
  return (u16)(u >> 16);
}

__device__ __forceinline__ v8f wmma_bf16(v16bf a, v16bf b, v8f c) {
  return __builtin_amdgcn_wmma_f32_16x16x32_bf16(false, a, false, b, (short)0, c,
                                                 false, false);
}

// ---------------------------------------------------------------- convert
__global__ void cvt_f32_bf16(const float* __restrict__ s, u16* __restrict__ d, int n) {
  int i = blockIdx.x * blockDim.x + threadIdx.x;
  if (i < n) d[i] = f2bf(s[i]);
}

// ------------------------------------------------- depthwise conv downsample
// xd[b, nk, c] = bias[c] + sum_t x[b, 2*nk-1+t, c] * w[c,0,t]   (zero pad)
__global__ void conv_ds_kernel(const float* __restrict__ x, const float* __restrict__ w,
                               const float* __restrict__ bias, u16* __restrict__ xd,
                               int B, int N, int C, int Nk) {
  int i = blockIdx.x * blockDim.x + threadIdx.x;
  int total = B * Nk * C;
  if (i >= total) return;
  int c  = i % C;
  int nk = (i / C) % Nk;
  int bb = i / (C * Nk);
  float acc = bias[c];
  int nbase = nk * 2 - 1;
#pragma unroll
  for (int t = 0; t < 3; t++) {
    int n = nbase + t;
    if (n >= 0 && n < N)
      acc += x[((size_t)bb * N + n) * C + c] * w[c * 3 + t];
  }
  xd[i] = f2bf(acc);
}

// ------------------------------------------------------------- WMMA GEMM
// out[m,n] = sum_k A[m,k]*W[n,k] + bias[n]
// wave tile 32x64: 2 A-frags x 4 B-frags -> 8 accumulators.
// Double-buffered k-loop: loads for step i+1 overlap WMMAs for step i.
// __launch_bounds__(256, 2): VGPR budget for ~200 live regs, no spills,
// while keeping 2 waves/SIMD for inter-wave latency hiding.
// mode 0: bf16 row-major [M,Nout]
// mode 1: bf16 transposed per-head: vt[((b*H+h)*64+d)*Nk + key]
// mode 2: f32 row-major [M,Nout]
struct GemmFrags { BF16x16 a0, a1, b[4]; };

__global__ void __launch_bounds__(256, 2)
gemm_bf16_wmma(const u16* __restrict__ A, const u16* __restrict__ W,
               const float* __restrict__ bias, void* __restrict__ out,
               int M, int K, int Nout, int mode, int Nk, int H) {
  const int lane = threadIdx.x & 31;
  const int wave = threadIdx.x >> 5;
  const int col  = lane & 15;
  const int half = lane >> 4;
  const int ntiles = Nout >> 6;
  int wid = blockIdx.x * 8 + wave;
  int mt  = wid / ntiles;
  int nt  = wid % ntiles;
  int m0  = mt * 32;
  int n0  = nt * 64;
  if (m0 >= M) return;

  v8f acc[2][4];
#pragma unroll
  for (int i = 0; i < 2; i++)
#pragma unroll
    for (int j = 0; j < 4; j++) acc[i][j] = (v8f){};

  const u16* arow0 = A + (size_t)(m0 + col) * K + 8 * half;
  const u16* arow1 = A + (size_t)(m0 + 16 + col) * K + 8 * half;
  const u16* wbase[4];
#pragma unroll
  for (int nf = 0; nf < 4; nf++)
    wbase[nf] = W + (size_t)(n0 + nf * 16 + col) * K + 8 * half;

  auto loadf = [&](int k0, GemmFrags& f) {
    f.a0.h[0] = *(const v8u16*)(arow0 + k0);
    f.a0.h[1] = *(const v8u16*)(arow0 + k0 + 16);
    f.a1.h[0] = *(const v8u16*)(arow1 + k0);
    f.a1.h[1] = *(const v8u16*)(arow1 + k0 + 16);
#pragma unroll
    for (int nf = 0; nf < 4; nf++) {
      f.b[nf].h[0] = *(const v8u16*)(wbase[nf] + k0);
      f.b[nf].h[1] = *(const v8u16*)(wbase[nf] + k0 + 16);
    }
  };
  auto comp = [&](GemmFrags& f) {
#pragma unroll
    for (int nf = 0; nf < 4; nf++) {
      acc[0][nf] = wmma_bf16(f.a0.v, f.b[nf].v, acc[0][nf]);
      acc[1][nf] = wmma_bf16(f.a1.v, f.b[nf].v, acc[1][nf]);
    }
  };

  GemmFrags fA, fB;
  loadf(0, fA);
  for (int k0 = 0; k0 + 64 <= K; k0 += 64) {
    loadf(k0 + 32, fB);                   // in flight while fA computes
    comp(fA);
    if (k0 + 64 < K) loadf(k0 + 64, fA);  // in flight while fB computes
    comp(fB);
  }

  if (mode == 0) {
    u16* o = (u16*)out;
#pragma unroll
    for (int i = 0; i < 2; i++)
#pragma unroll
      for (int nf = 0; nf < 4; nf++) {
        int n = n0 + nf * 16 + col;
        float bv = bias[n];
#pragma unroll
        for (int r = 0; r < 8; r++) {
          int m = m0 + i * 16 + 8 * half + r;
          o[(size_t)m * Nout + n] = f2bf(acc[i][nf][r] + bv);
        }
      }
  } else if (mode == 2) {
    float* o = (float*)out;
#pragma unroll
    for (int i = 0; i < 2; i++)
#pragma unroll
      for (int nf = 0; nf < 4; nf++) {
        int n = n0 + nf * 16 + col;
        float bv = bias[n];
#pragma unroll
        for (int r = 0; r < 8; r++) {
          int m = m0 + i * 16 + 8 * half + r;
          o[(size_t)m * Nout + n] = acc[i][nf][r] + bv;
        }
      }
  } else {  // mode 1: vt[((b*H+h)*64+d)*Nk + key], 8 keys contiguous -> 16B store
    u16* o = (u16*)out;
#pragma unroll
    for (int i = 0; i < 2; i++) {
      int keyb = m0 + i * 16 + 8 * half;
      int bidx = keyb / Nk;
      int key  = keyb % Nk;
#pragma unroll
      for (int nf = 0; nf < 4; nf++) {
        int n = n0 + nf * 16 + col;
        float bv = bias[n];
        int hh = n >> 6, dd = n & 63;
        v8u16 pack;
#pragma unroll
        for (int r = 0; r < 8; r++) pack[r] = f2bf(acc[i][nf][r] + bv);
        *(v8u16*)(o + (((size_t)bidx * H + hh) * 64 + dd) * Nk + key) = pack;
      }
    }
  }
}

// --------------------------------------------------------- flash attention
// one wave: 16 q-rows x D=64, streaming keys in chunks of 32, online softmax.
// K-fragments double-buffered; V-fragments issued before the softmax/LDS phase.
#define ATT_WAVES 8
struct KFrags { BF16x16 f[4]; };  // [sf*2 + kd]

__global__ void __launch_bounds__(256, 2)
attn_kernel(const u16* __restrict__ Q, const u16* __restrict__ Kb,
            const u16* __restrict__ Vt, u16* __restrict__ Oo,
            int B, int H, int N, int Nk, float scale) {
  __shared__ __align__(16) u16 plds[ATT_WAVES * 16 * 32];
  const int lane = threadIdx.x & 31;
  const int wave = threadIdx.x >> 5;
  const int col  = lane & 15;
  const int half = lane >> 4;
  const int C = H * 64;
  const int qtiles = N >> 4;
  int wid = blockIdx.x * ATT_WAVES + wave;
  int bh = wid / qtiles;
  int mt = wid % qtiles;
  int b = bh / H;
  int h = bh % H;
  int m0 = mt * 16;
  u16* my = plds + wave * 512;

  // Q A-frags (d-halves 0..31, 32..63)
  BF16x16 qf[2];
  const u16* qrow = Q + ((size_t)(b * N + m0 + col)) * C + h * 64;
#pragma unroll
  for (int kd = 0; kd < 2; kd++) {
    int base = kd * 32 + 16 * half;
    qf[kd].h[0] = *(const v8u16*)(qrow + base);
    qf[kd].h[1] = *(const v8u16*)(qrow + base + 8);
  }

  v8f o[4];
#pragma unroll
  for (int f = 0; f < 4; f++) o[f] = (v8f){};
  float mrow[8], lrow[8];
#pragma unroll
  for (int r = 0; r < 8; r++) { mrow[r] = -1e30f; lrow[r] = 0.f; }

  const u16* kbase = Kb + (size_t)b * Nk * C + h * 64 + 16 * half;
  const u16* vbase = Vt + ((size_t)(bh * 64 + col)) * Nk + 16 * half;

  auto loadK = [&](int key0, KFrags& kf) {
#pragma unroll
    for (int sf = 0; sf < 2; sf++) {
      const u16* kp = kbase + (size_t)(key0 + sf * 16 + col) * C;
      kf.f[sf * 2 + 0].h[0] = *(const v8u16*)(kp);
      kf.f[sf * 2 + 0].h[1] = *(const v8u16*)(kp + 8);
      kf.f[sf * 2 + 1].h[0] = *(const v8u16*)(kp + 32);
      kf.f[sf * 2 + 1].h[1] = *(const v8u16*)(kp + 40);
    }
  };

  auto stage = [&](int key0, KFrags& cur, KFrags& nxt, bool pref) {
    // S = scale * Q @ K^T for 32 keys (two 16-wide frags)
    v8f s[2];
#pragma unroll
    for (int sf = 0; sf < 2; sf++) {
      v8f z = (v8f){};
      z = wmma_bf16(qf[0].v, cur.f[sf * 2 + 0].v, z);
      z = wmma_bf16(qf[1].v, cur.f[sf * 2 + 1].v, z);
      s[sf] = z;
    }
    // prefetch next chunk's K while softmax runs
    if (pref) loadK(key0 + 32, nxt);
    // issue V loads now; consumed after the LDS round-trip
    BF16x16 vf[4];
#pragma unroll
    for (int df = 0; df < 4; df++) {
      const u16* vp = vbase + (size_t)(df * 16) * Nk + key0;
      vf[df].h[0] = *(const v8u16*)(vp);
      vf[df].h[1] = *(const v8u16*)(vp + 8);
    }

    float mx[8], p0a[8], p1a[8], rs[8];
#pragma unroll
    for (int r = 0; r < 8; r++) {
      s[0][r] *= scale;
      s[1][r] *= scale;
      mx[r] = fmaxf(s[0][r], s[1][r]);
    }
#pragma unroll
    for (int off = 1; off < 16; off <<= 1)
#pragma unroll
      for (int r = 0; r < 8; r++)
        mx[r] = fmaxf(mx[r], __shfl_xor(mx[r], off, 32));
#pragma unroll
    for (int r = 0; r < 8; r++) {
      float mnew  = fmaxf(mrow[r], mx[r]);
      float alpha = __expf(mrow[r] - mnew);
      mrow[r] = mnew;
      p0a[r] = __expf(s[0][r] - mnew);
      p1a[r] = __expf(s[1][r] - mnew);
      rs[r]  = p0a[r] + p1a[r];
      lrow[r] *= alpha;
#pragma unroll
      for (int f = 0; f < 4; f++) o[f][r] *= alpha;
    }
#pragma unroll
    for (int off = 1; off < 16; off <<= 1)
#pragma unroll
      for (int r = 0; r < 8; r++)
        rs[r] += __shfl_xor(rs[r], off, 32);
#pragma unroll
    for (int r = 0; r < 8; r++) lrow[r] += rs[r];

    // P: C-frag -> LDS -> A-frag
#pragma unroll
    for (int r = 0; r < 8; r++) {
      int row = 8 * half + r;
      my[row * 32 + col]      = f2bf(p0a[r]);
      my[row * 32 + 16 + col] = f2bf(p1a[r]);
    }
    __syncthreads();
    BF16x16 pf;
    {
      const u16* pr = my + col * 32 + 16 * half;
      pf.h[0] = *(const v8u16*)(pr);
      pf.h[1] = *(const v8u16*)(pr + 8);
    }

    // O += P @ V
#pragma unroll
    for (int df = 0; df < 4; df++)
      o[df] = wmma_bf16(pf.v, vf[df].v, o[df]);
    __syncthreads();
  };

  KFrags kA, kB;
  loadK(0, kA);
  for (int key0 = 0; key0 < Nk; key0 += 64) {
    stage(key0, kA, kB, true);
    stage(key0 + 32, kB, kA, key0 + 64 < Nk);
  }

  // normalize and store bf16 [B,N,C]
  const size_t obase = ((size_t)(b * N + m0 + 8 * half)) * C + h * 64;
#pragma unroll
  for (int df = 0; df < 4; df++)
#pragma unroll
    for (int r = 0; r < 8; r++) {
      float v = o[df][r] / lrow[r];
      Oo[obase + (size_t)r * C + df * 16 + col] = f2bf(v);
    }
}

// ------------------------------------------------------------------ launch
extern "C" void kernel_launch(void* const* d_in, const int* in_sizes, int n_in,
                              void* d_out, int out_size, void* d_ws, size_t ws_size,
                              hipStream_t stream) {
  (void)in_sizes; (void)n_in; (void)out_size; (void)ws_size;
  const int B = 2, N = 4096, C = 1024, H = 16;
  const int Nk = N / 2;

  const float* x      = (const float*)d_in[0];
  const float* q_w    = (const float*)d_in[1];
  const float* q_b    = (const float*)d_in[2];
  const float* k_w    = (const float*)d_in[3];
  const float* k_b    = (const float*)d_in[4];
  const float* v_w    = (const float*)d_in[5];
  const float* v_b    = (const float*)d_in[6];
  const float* conv_w = (const float*)d_in[7];
  const float* conv_b = (const float*)d_in[8];
  const float* out_w  = (const float*)d_in[9];
  const float* out_b  = (const float*)d_in[10];

  char* ws = (char*)d_ws;
  size_t off = 0;
  auto alloc = [&](size_t bytes) -> void* {
    void* p = ws + off;
    off += (bytes + 255) & ~(size_t)255;
    return p;
  };
  u16* x_bf  = (u16*)alloc((size_t)B * N * C * 2);
  u16* q_bf  = (u16*)alloc((size_t)B * N * C * 2);
  u16* xd_bf = (u16*)alloc((size_t)B * Nk * C * 2);
  u16* k_bf  = (u16*)alloc((size_t)B * Nk * C * 2);
  u16* vt_bf = (u16*)alloc((size_t)B * H * 64 * Nk * 2);
  u16* at_bf = (u16*)alloc((size_t)B * N * C * 2);
  u16* wq    = (u16*)alloc((size_t)C * C * 2);
  u16* wk    = (u16*)alloc((size_t)C * C * 2);
  u16* wv    = (u16*)alloc((size_t)C * C * 2);
  u16* wo    = (u16*)alloc((size_t)C * C * 2);

  int nx = B * N * C;
  int nw = C * C;
  cvt_f32_bf16<<<(nx + 255) / 256, 256, 0, stream>>>(x, x_bf, nx);
  cvt_f32_bf16<<<(nw + 255) / 256, 256, 0, stream>>>(q_w, wq, nw);
  cvt_f32_bf16<<<(nw + 255) / 256, 256, 0, stream>>>(k_w, wk, nw);
  cvt_f32_bf16<<<(nw + 255) / 256, 256, 0, stream>>>(v_w, wv, nw);
  cvt_f32_bf16<<<(nw + 255) / 256, 256, 0, stream>>>(out_w, wo, nw);

  int nconv = B * Nk * C;
  conv_ds_kernel<<<(nconv + 255) / 256, 256, 0, stream>>>(x, conv_w, conv_b, xd_bf,
                                                          B, N, C, Nk);

  auto gemm = [&](const u16* A, const u16* W, const float* bias, void* o, int M, int mode) {
    int blocks = (M / 32) * (C / 64) / 8;
    gemm_bf16_wmma<<<blocks, 256, 0, stream>>>(A, W, bias, o, M, C, C, mode, Nk, H);
  };
  gemm(x_bf,  wq, q_b, q_bf,  B * N,  0);   // q = x @ q_w^T + q_b
  gemm(xd_bf, wk, k_b, k_bf,  B * Nk, 0);   // k
  gemm(xd_bf, wv, v_b, vt_bf, B * Nk, 1);   // v -> transposed per head

  int ablocks = (B * H * (N / 16)) / ATT_WAVES;  // 1024
  attn_kernel<<<ablocks, 256, 0, stream>>>(q_bf, k_bf, vt_bf, at_bf, B, H, N, Nk, 0.125f);

  gemm(at_bf, wo, out_b, (float*)d_out, B * N, 2);  // final projection, f32 out
}
